// Decoder_3745211482823
// MI455X (gfx1250) — matrix operand order repaired and checked by reading, not compiled, not run on hardware
//
#include <hip/hip_runtime.h>
#include <hip/hip_bf16.h>
#include <stdint.h>

// ---------------------------------------------------------------------------
// Types / WMMA helpers (CDNA5 wave32, v_wmma_f32_16x16x32_bf16)
// ---------------------------------------------------------------------------
typedef __bf16 bf16;
typedef __attribute__((ext_vector_type(16))) __bf16 v16bf;
typedef __attribute__((ext_vector_type(8)))  __bf16 v8bf;
typedef __attribute__((ext_vector_type(8)))  float  v8f;

#define DEV static __device__ __forceinline__

DEV float sigmoidf(float x) { return 1.0f / (1.0f + __expf(-x)); }

// A fragment: 16x32 bf16, row-major source, lda in elements.
// ISA 7.12.2: lane<16 -> row=lane, K 0..7 & 16..23 ; lane>=16 -> K 8..15 & 24..31
DEV v16bf ld_a16x32(const bf16* A, int lda, int m0, int k0) {
  const int lane = threadIdx.x & 31;
  const bf16* p = A + (size_t)(m0 + (lane & 15)) * lda + (size_t)(k0 + ((lane >> 4) << 3));
  v8bf lo = *(const v8bf*)p;
  v8bf hi = *(const v8bf*)(p + 16);
  v16bf r;
#pragma unroll
  for (int i = 0; i < 8; ++i) { r[i] = lo[i]; r[8 + i] = hi[i]; }
  return r;
}

// B fragment from N x K row-major (packed/transposed weight, C = A * W^T).
// All B buffers are padded to N multiple of 16, so no guards: two b128 loads.
// lanes 0..15 hold K k0..k0+7 & k0+16..k0+23 of column n, lanes 16..31 hold K k0+8..k0+15 & k0+24..k0+31.
DEV v16bf ld_b_nk(const bf16* W, int ldw, int k0, int n0) {
  const int lane = threadIdx.x & 31;
  const int n = n0 + (lane & 15);
  const bf16* p = W + (size_t)n * ldw + (size_t)(k0 + ((lane >> 4) << 3));
  v8bf lo = *(const v8bf*)p;
  v8bf hi = *(const v8bf*)(p + 16);
  v16bf r;
#pragma unroll
  for (int i = 0; i < 8; ++i) { r[i] = lo[i]; r[8 + i] = hi[i]; }
  return r;
}

DEV v8f wmma_bf16(v16bf a, v16bf b, v8f c) {
  return __builtin_amdgcn_wmma_f32_16x16x32_bf16(false, a, false, b, (short)0, c,
                                                 false, false);
}

// ---------------------------------------------------------------------------
// Generic bf16 GEMM: C(MxN) = A(MxK) * W^T (+bias, optional relu).
// A: MxK row-major (M multiple of 16, zero-padded rows OK).
// B: N_pad x K row-major packed bf16 (N padded to multiple of 16, zero rows).
// K multiple of 32. Grid: x = ceil(N/128) (8 waves * 16 cols), y = M/16.
// ---------------------------------------------------------------------------
__global__ void __launch_bounds__(256) k_gemm(const bf16* __restrict__ A,
                                              const bf16* __restrict__ B,
                                              float* __restrict__ C,
                                              bf16* __restrict__ Cbf,
                                              const float* __restrict__ bias,
                                              int M, int N, int K,
                                              int lda, int ldc, int relu) {
  const int lane = threadIdx.x & 31;
  const int wv = threadIdx.x >> 5;
  const int n0 = (blockIdx.x * 8 + wv) * 16;
  const int m0 = blockIdx.y * 16;
  if (n0 >= N) return;
  v8f acc = {};
#pragma unroll 2
  for (int k = 0; k < K; k += 32) {
    v16bf a = ld_a16x32(A, lda, m0, k);
    v16bf b = ld_b_nk(B, K, k, n0);
    acc = wmma_bf16(a, b, acc);
  }
  const int n = n0 + (lane & 15);
  if (n >= N) return;
  const float bv = bias ? bias[n] : 0.0f;
  const int mb = (lane >> 4) << 3;
#pragma unroll
  for (int j = 0; j < 8; ++j) {
    int m = m0 + mb + j;
    if (m < M) {
      float v = acc[j] + bv;
      if (relu) v = fmaxf(v, 0.0f);
      if (C)   C[(size_t)m * ldc + n] = v;
      if (Cbf) Cbf[(size_t)m * ldc + n] = (bf16)v;
    }
  }
}

// ---------------------------------------------------------------------------
// Small utility kernels
// ---------------------------------------------------------------------------
__global__ void k_zero8(uint8_t* p, size_t n) {
  size_t i = (size_t)blockIdx.x * 256 + threadIdx.x;
  if (i < n) p[i] = 0;
}
// Transpose f32 (K,N) -> packed bf16 (Npad,K); rows n>=N are zero.
__global__ void k_transpose_f2b(const float* __restrict__ s, bf16* __restrict__ d,
                                int K, int N, int Npad) {
  int i = blockIdx.x * 256 + threadIdx.x;
  if (i >= Npad * K) return;
  int k = i % K, n = i / K;
  d[(size_t)n * K + k] = (n < N) ? (bf16)s[(size_t)k * N + n] : (bf16)0.0f;
}
__global__ void k_f2b(const float* __restrict__ s, bf16* __restrict__ d, int n) {
  int i = blockIdx.x * 256 + threadIdx.x;
  if (i < n) d[i] = (bf16)s[i];
}
// Copy f32 (rows, cols) sub-block -> bf16 at (dld,dc0) (used to pack NxK weights).
__global__ void k_copy2d_f2b(const float* __restrict__ s, int sld, int sc0,
                             bf16* __restrict__ d, int dld, int dc0,
                             int rows, int cols) {
  int i = blockIdx.x * 256 + threadIdx.x;
  if (i >= rows * cols) return;
  int r = i / cols, c = i % cols;
  d[(size_t)r * dld + dc0 + c] = (bf16)s[(size_t)r * sld + sc0 + c];
}
__global__ void k_addv(const float* a, const float* b, float* d, int n) {
  int i = blockIdx.x * 256 + threadIdx.x;
  if (i < n) d[i] = a[i] + b[i];
}

// prev_out_seq rows (r = td*8+b) in bf16: td==0 -> 0, else ft[b][td-1][cr]
// ft[b][j][cr] = feature_target[b][cr%80][2*j + cr/80]
__global__ void k_build_prenet_in(const float* __restrict__ ft, bf16* __restrict__ A) {
  int i = blockIdx.x * 256 + threadIdx.x;
  if (i >= 400 * 160) return;
  int cr = i % 160, r = i / 160;
  int td = r / 8, b = r % 8;
  float v = 0.0f;
  if (td > 0) {
    int c = cr % 80, q = cr / 80;
    v = ft[((size_t)b * 80 + c) * 100 + 2 * (td - 1) + q];
  }
  A[i] = (bf16)v;
}

// prev_lip images (400,1,48,48) f32; lip_target layout (8,1,48,48,50)
__global__ void k_build_lipin(const float* __restrict__ lt, float* __restrict__ y) {
  int i = blockIdx.x * 256 + threadIdx.x;
  if (i >= 400 * 48 * 48) return;
  int pix = i % 2304, r = i / 2304;
  int td = r / 8, b = r % 8;
  int h = pix / 48, w = pix % 48;
  float v = 0.0f;
  if (td > 0) v = lt[(((size_t)b * 48 + h) * 48 + w) * 50 + (td - 1)];
  y[i] = v;
}

// ---------------------------------------------------------------------------
// Direct convolution kernels (f32, NCHW). fuse: BN(g,be)+ReLU after bias.
// ---------------------------------------------------------------------------
__global__ void k_conv3(const float* __restrict__ x, const float* __restrict__ wgt,
                        const float* __restrict__ bias, const float* __restrict__ g,
                        const float* __restrict__ be, float* __restrict__ y,
                        int N, int Ci, int Co, int H, int W, int fuse) {
  int idx = blockIdx.x * 256 + threadIdx.x;
  int total = N * Co * H * W;
  if (idx >= total) return;
  int wq = idx % W; int t = idx / W;
  int hp = t % H; t /= H;
  int o = t % Co; int n = t / Co;
  const float* xb = x + (size_t)n * Ci * H * W;
  const float* wb = wgt + (size_t)o * Ci * 9;
  float s = bias[o];
  for (int ci = 0; ci < Ci; ++ci) {
    const float* xc = xb + (size_t)ci * H * W;
    const float* wc = wb + ci * 9;
#pragma unroll
    for (int kh = 0; kh < 3; ++kh) {
      int ih = hp + kh - 1;
      if (ih < 0 || ih >= H) continue;
#pragma unroll
      for (int kw = 0; kw < 3; ++kw) {
        int iw = wq + kw - 1;
        if (iw < 0 || iw >= W) continue;
        s += xc[ih * W + iw] * wc[kh * 3 + kw];
      }
    }
  }
  if (fuse) s = fmaxf(s * g[o] + be[o], 0.0f);
  y[idx] = s;
}

// conv 4x4 stride 2 pad 1, weights (Co,Ci,4,4). Output (N,Co,H/2,W/2).
__global__ void k_conv4s2(const float* __restrict__ x, const float* __restrict__ wgt,
                          const float* __restrict__ bias, const float* __restrict__ g,
                          const float* __restrict__ be, float* __restrict__ y,
                          int N, int Ci, int Co, int H, int W, int fuse) {
  int Ho = H >> 1, Wo = W >> 1;
  int idx = blockIdx.x * 256 + threadIdx.x;
  int total = N * Co * Ho * Wo;
  if (idx >= total) return;
  int wq = idx % Wo; int t = idx / Wo;
  int hp = t % Ho; t /= Ho;
  int o = t % Co; int n = t / Co;
  const float* xb = x + (size_t)n * Ci * H * W;
  const float* wb = wgt + (size_t)o * Ci * 16;
  float s = bias[o];
  for (int ci = 0; ci < Ci; ++ci) {
    const float* xc = xb + (size_t)ci * H * W;
    const float* wc = wb + ci * 16;
#pragma unroll
    for (int kh = 0; kh < 4; ++kh) {
      int ih = 2 * hp - 1 + kh;
      if (ih < 0 || ih >= H) continue;
#pragma unroll
      for (int kw = 0; kw < 4; ++kw) {
        int iw = 2 * wq - 1 + kw;
        if (iw < 0 || iw >= W) continue;
        s += xc[ih * W + iw] * wc[kh * 4 + kw];
      }
    }
  }
  if (fuse) s = fmaxf(s * g[o] + be[o], 0.0f);
  y[idx] = s;
}

// ConvTranspose2d k=4 stride=2 pad=1, torch weights (Ci,Co,4,4). Out (N,Co,2H,2W).
__global__ void k_convT4s2(const float* __restrict__ x, const float* __restrict__ wgt,
                           const float* __restrict__ bias, const float* __restrict__ g,
                           const float* __restrict__ be, float* __restrict__ y,
                           int N, int Ci, int Co, int H, int W, int fuse) {
  int Ho = H << 1, Wo = W << 1;
  int idx = blockIdx.x * 256 + threadIdx.x;
  int total = N * Co * Ho * Wo;
  if (idx >= total) return;
  int wq = idx % Wo; int t = idx / Wo;
  int hp = t % Ho; t /= Ho;
  int o = t % Co; int n = t / Co;
  const float* xb = x + (size_t)n * Ci * H * W;
  float s = bias[o];
  for (int ci = 0; ci < Ci; ++ci) {
    const float* xc = xb + (size_t)ci * H * W;
    const float* wc = wgt + ((size_t)ci * Co + o) * 16;
#pragma unroll
    for (int kh = 0; kh < 4; ++kh) {
      int th = hp + 1 - kh;
      if (th < 0 || (th & 1)) continue;
      int ih = th >> 1;
      if (ih >= H) continue;
#pragma unroll
      for (int kw = 0; kw < 4; ++kw) {
        int tw = wq + 1 - kw;
        if (tw < 0 || (tw & 1)) continue;
        int iw = tw >> 1;
        if (iw >= W) continue;
        s += xc[ih * W + iw] * wc[kh * 4 + kw];
      }
    }
  }
  if (fuse) s = fmaxf(s * g[o] + be[o], 0.0f);
  y[idx] = s;
}

// mean over 3x3 of (400,256,3,3) -> bf16 into Xcat[:,256:512]
__global__ void k_meanpool(const float* __restrict__ x, bf16* __restrict__ xcat) {
  int i = blockIdx.x * 256 + threadIdx.x;
  if (i >= 400 * 256) return;
  int c = i % 256, r = i / 256;
  const float* p = x + ((size_t)r * 256 + c) * 9;
  float s = 0.0f;
#pragma unroll
  for (int j = 0; j < 9; ++j) s += p[j];
  xcat[(size_t)r * 512 + 256 + c] = (bf16)(s * (1.0f / 9.0f));
}

// broadcast (400,128) -> (400,128,3,3)
__global__ void k_bcast9(const float* __restrict__ lin, float* __restrict__ y) {
  int i = blockIdx.x * 256 + threadIdx.x;
  if (i >= 400 * 128 * 9) return;
  y[i] = lin[i / 9];
}

// ---------------------------------------------------------------------------
// Output re-layout kernels
// ---------------------------------------------------------------------------
__global__ void k_out_feat(const float* __restrict__ feats, float* __restrict__ out) {
  int i = blockIdx.x * 256 + threadIdx.x;
  if (i >= 8 * 80 * 100) return;
  int t2 = i % 100; int t = i / 100;
  int c = t % 80; int b = t / 80;
  int td = t2 >> 1, q = t2 & 1;
  out[i] = feats[((size_t)td * 8 + b) * 160 + q * 80 + c];
}
__global__ void k_out_logit(const float* __restrict__ probs, float* __restrict__ out) {
  int i = blockIdx.x * 256 + threadIdx.x;
  if (i >= 800) return;
  int t2 = i % 100, b = i / 100;
  int td = t2 >> 1, q = t2 & 1;
  out[i] = probs[((size_t)td * 8 + b) * 2 + q];
}
__global__ void k_out_lip(const float* __restrict__ lip, float* __restrict__ out) {
  int i = blockIdx.x * 256 + threadIdx.x;
  if (i >= 8 * 48 * 48 * 50) return;
  int td = i % 50; int t = i / 50;
  int w = t % 48; t /= 48;
  int h = t % 48; int b = t / 48;
  out[i] = lip[((size_t)td * 8 + b) * 2304 + h * 48 + w];
}

// ---------------------------------------------------------------------------
// Persistent sequential decoder (attention + 2 zoneout-LSTMs, 50 steps).
// 32 workgroups x 256 threads; device-scope generation barrier in d_ws.
// ---------------------------------------------------------------------------
#define NWG 32

struct DecArgs {
  const float* enc;        // (8,150,256)
  const float* enc_proj;   // (1200,128) = enc@W_enc + b_enc
  const int*   text_len;   // (8)
  const float* conv_w;     // (32,31)
  const float* W_conv;     // (32,128)
  const float* vvec;       // (128)
  const bf16*  Wdec;       // (128,1024) NxK packed (transposed W_dec)
  const bf16*  Wcomb1;     // (4096,1280) NxK = [Wih1[:, :256] | Whh1]
  const bf16*  Wcomb2;     // (4096,2048) NxK = [Wih2 | Whh2]
  const float* xpart;      // (400,4096) = x@Wih1[:,256:]^T + bih1 + bhh1
  const float* gbias2;     // (4096) = bih2 + bhh2
  bf16* A1;                // (16,1280) bf16 = [att_c | h0], rows 8..15 zero
  bf16* A2;                // (16,2048) bf16 = [h0n | h1]
  float* h0; float* c0; float* h1; float* c1;  // (8,1024) each
  float* cum;              // (8,150)
  float* ebuf;             // (8,150)
  bf16* hcs;               // (400,1280) bf16 = [h1n | att_c] per (step,b)
  float* att_out;          // (8,50,150) -> final output region 4
  int* bar;                // [0]=count [1]=generation
};

DEV void gsync(int* bar) {
  __syncthreads();
  if (threadIdx.x == 0) {
    __threadfence();
    int g = __hip_atomic_load(&bar[1], __ATOMIC_ACQUIRE, __HIP_MEMORY_SCOPE_AGENT);
    int a = __hip_atomic_fetch_add(&bar[0], 1, __ATOMIC_ACQ_REL, __HIP_MEMORY_SCOPE_AGENT);
    if (a == NWG - 1) {
      __hip_atomic_store(&bar[0], 0, __ATOMIC_RELAXED, __HIP_MEMORY_SCOPE_AGENT);
      __hip_atomic_fetch_add(&bar[1], 1, __ATOMIC_ACQ_REL, __HIP_MEMORY_SCOPE_AGENT);
    } else {
      while (__hip_atomic_load(&bar[1], __ATOMIC_ACQUIRE, __HIP_MEMORY_SCOPE_AGENT) == g)
        __builtin_amdgcn_s_sleep(2);
    }
  }
  __syncthreads();
}

__global__ void __launch_bounds__(256, 1) k_decoder(DecArgs p) {
  __shared__ float s_prev[160];       // attention weights scratch
  __shared__ float s_cf[32][152];     // location conv features
  __shared__ float s_dec[128];        // h0 @ W_dec for this batch row
  __shared__ float s_g[4][2][16][17]; // gate tiles for recombine
  __shared__ float s_red[256];

  const int tid = threadIdx.x, lane = tid & 31, wv = tid >> 5;
  const int wg = blockIdx.x;

  for (int step = 0; step < 50; ++step) {
    // ------------------ Phase 1: dec_proj + location conv + energies
    {
      const int b = wg >> 2, part = wg & 3;
      {
        v8f acc = {};
        const int n0 = wv * 16;
#pragma unroll 2
        for (int k = 0; k < 1024; k += 32) {
          v16bf a = ld_a16x32(p.A1, 1280, 0, 256 + k);
          v16bf bb = ld_b_nk(p.Wdec, 1024, k, n0);
          acc = wmma_bf16(a, bb, acc);
        }
        const int n = n0 + (lane & 15);
        const int mb = (lane >> 4) << 3;
#pragma unroll
        for (int j = 0; j < 8; ++j)
          if (mb + j == b) s_dec[n] = acc[j];
      }
      const int len = p.text_len[b];
      for (int t = tid; t < 150; t += 256) {
        s_prev[t] = (step == 0) ? ((t < len) ? 1.0f / (float)len : 0.0f)
                                : p.cum[b * 150 + t];
      }
      __syncthreads();
      for (int i = tid; i < 32 * 150; i += 256) {
        int c = i / 150, t = i % 150;
        float s = 0.0f;
#pragma unroll
        for (int k = 0; k < 31; ++k) {
          int tt = t - 15 + k;
          if (tt >= 0 && tt < 150) s += s_prev[tt] * p.conv_w[c * 31 + k];
        }
        s_cf[c][t] = s;
      }
      __syncthreads();
      int t0 = part * 38, t1 = t0 + 38; if (t1 > 150) t1 = 150;
      for (int t = t0 + wv; t < t1; t += 8) {
        float acc = 0.0f;
        for (int h = lane; h < 128; h += 32) {
          float s = p.enc_proj[(size_t)(b * 150 + t) * 128 + h] + s_dec[h];
#pragma unroll
          for (int c = 0; c < 32; ++c) s += s_cf[c][t] * p.W_conv[c * 128 + h];
          acc += p.vvec[h] * tanhf(s);
        }
#pragma unroll
        for (int off = 16; off > 0; off >>= 1) acc += __shfl_xor(acc, off, 32);
        if (lane == 0) p.ebuf[b * 150 + t] = (t < len) ? acc : -1e9f;
      }
    }
    gsync(p.bar);

    // ------------------ Phase 1b: softmax + context (8 WGs, one per batch)
    if (wg < 8) {
      const int b = wg;
      float mx = -1e30f;
      for (int t = tid; t < 150; t += 256) mx = fmaxf(mx, p.ebuf[b * 150 + t]);
      s_red[tid] = mx; __syncthreads();
      for (int s2 = 128; s2 > 0; s2 >>= 1) {
        if (tid < s2) s_red[tid] = fmaxf(s_red[tid], s_red[tid + s2]);
        __syncthreads();
      }
      mx = s_red[0]; __syncthreads();
      float psum = 0.0f;
      for (int t = tid; t < 150; t += 256) {
        float ex = __expf(p.ebuf[b * 150 + t] - mx);
        s_prev[t] = ex; psum += ex;
      }
      s_red[tid] = psum; __syncthreads();
      for (int s2 = 128; s2 > 0; s2 >>= 1) {
        if (tid < s2) s_red[tid] += s_red[tid + s2];
        __syncthreads();
      }
      const float inv = 1.0f / s_red[0]; __syncthreads();
      for (int t = tid; t < 150; t += 256) {
        float aw = s_prev[t] * inv;
        s_prev[t] = aw;
        p.att_out[((size_t)b * 50 + step) * 150 + t] = aw;
        p.cum[b * 150 + t] += aw;
      }
      __syncthreads();
      {
        const int ch = tid;  // 0..255
        float s = 0.0f;
        for (int t = 0; t < 150; ++t)
          s += s_prev[t] * p.enc[((size_t)b * 150 + t) * 256 + ch];
        p.A1[(size_t)b * 1280 + ch] = (bf16)s;
        p.hcs[((size_t)step * 8 + b) * 1280 + 1024 + ch] = (bf16)s;
      }
    }
    gsync(p.bar);

    // ------------------ Phase 2: LSTM1 gates (K=1280) + zoneout update
    {
      const int u0 = wg * 32;
      const int gate = wv >> 1, ct = wv & 1;
      const int n0 = gate * 1024 + u0 + ct * 16;
      const int nn = n0 + (lane & 15);
      const int mb = (lane >> 4) << 3;
      v8f acc;
#pragma unroll
      for (int j = 0; j < 8; ++j) {
        int m = mb + j;
        acc[j] = (m < 8) ? p.xpart[((size_t)step * 8 + m) * 4096 + nn] : 0.0f;
      }
#pragma unroll 2
      for (int k = 0; k < 1280; k += 32) {
        __builtin_prefetch(p.Wcomb1 + (size_t)n0 * 1280 + k + 64, 0, 1);
        v16bf a = ld_a16x32(p.A1, 1280, 0, k);
        v16bf bb = ld_b_nk(p.Wcomb1, 1280, k, n0);
        acc = wmma_bf16(a, bb, acc);
      }
#pragma unroll
      for (int j = 0; j < 8; ++j) s_g[gate][ct][mb + j][lane & 15] = acc[j];
      __syncthreads();
    }
    gsync(p.bar);  // all WGs done reading A1 before h0 columns are rewritten
    {
      const int u0 = wg * 32;
      const int m = tid >> 5, u = tid & 31;
      const int c2 = u >> 4, uu = u & 15;
      float gi = s_g[0][c2][m][uu], gf = s_g[1][c2][m][uu];
      float gg = s_g[2][c2][m][uu], go = s_g[3][c2][m][uu];
      const int gu = u0 + u;
      float cold = p.c0[m * 1024 + gu], hold = p.h0[m * 1024 + gu];
      float cn = sigmoidf(gf) * cold + sigmoidf(gi) * tanhf(gg);
      float hn = sigmoidf(go) * tanhf(cn);
      cn = 0.1f * cold + 0.9f * cn;
      hn = 0.1f * hold + 0.9f * hn;
      p.c0[m * 1024 + gu] = cn;
      p.h0[m * 1024 + gu] = hn;
      p.A1[(size_t)m * 1280 + 256 + gu] = (bf16)hn;
      p.A2[(size_t)m * 2048 + gu] = (bf16)hn;
    }
    gsync(p.bar);

    // ------------------ Phase 3: LSTM2 gates (K=2048) + zoneout update
    {
      const int u0 = wg * 32;
      const int gate = wv >> 1, ct = wv & 1;
      const int n0 = gate * 1024 + u0 + ct * 16;
      const int nn = n0 + (lane & 15);
      const float bv = p.gbias2[nn];
      const int mb = (lane >> 4) << 3;
      v8f acc;
#pragma unroll
      for (int j = 0; j < 8; ++j) acc[j] = bv;
#pragma unroll 2
      for (int k = 0; k < 2048; k += 32) {
        __builtin_prefetch(p.Wcomb2 + (size_t)n0 * 2048 + k + 64, 0, 1);
        v16bf a = ld_a16x32(p.A2, 2048, 0, k);
        v16bf bb = ld_b_nk(p.Wcomb2, 2048, k, n0);
        acc = wmma_bf16(a, bb, acc);
      }
#pragma unroll
      for (int j = 0; j < 8; ++j) s_g[gate][ct][mb + j][lane & 15] = acc[j];
      __syncthreads();
    }
    gsync(p.bar);  // all WGs done reading A2 before h1 columns are rewritten
    {
      const int u0 = wg * 32;
      const int m = tid >> 5, u = tid & 31;
      const int c2 = u >> 4, uu = u & 15;
      float gi = s_g[0][c2][m][uu], gf = s_g[1][c2][m][uu];
      float gg = s_g[2][c2][m][uu], go = s_g[3][c2][m][uu];
      const int gu = u0 + u;
      float cold = p.c1[m * 1024 + gu], hold = p.h1[m * 1024 + gu];
      float cn = sigmoidf(gf) * cold + sigmoidf(gi) * tanhf(gg);
      float hn = sigmoidf(go) * tanhf(cn);
      cn = 0.1f * cold + 0.9f * cn;
      hn = 0.1f * hold + 0.9f * hn;
      p.c1[m * 1024 + gu] = cn;
      p.h1[m * 1024 + gu] = hn;
      p.A2[(size_t)m * 2048 + 1024 + gu] = (bf16)hn;
      p.hcs[((size_t)step * 8 + m) * 1280 + gu] = (bf16)hn;
    }
    gsync(p.bar);
  }
}

// ---------------------------------------------------------------------------
// Host orchestration
// ---------------------------------------------------------------------------
// Input flattening assumption: top-level setup_inputs() insertion order
// (params, enc_output, text_len, feature_target, lip_target); inside params,
// JAX leaf order (dict keys sorted alphabetically, lists by index).
enum {
  I_ATT_WCONVH = 0, I_ATT_WDEC = 1, I_ATT_WENC = 2, I_ATT_BENC = 3,
  I_ATT_CONVW = 4, I_ATT_V = 5,
  I_CAT_W = 6, I_CAT_B = 7,
  I_FEAT_W = 8,
  I_LO_BASE = 9,            // lip_out.blocks: 4 blocks x {b1,b2,be1,be2,g1,g2,w1,w2}
  I_LO_LINW = 41, I_LO_LINB = 42,
  I_LP_BASE = 43,           // lip_prenet: 4 blocks x 8 leaves
  I_L0_WHH = 75, I_L0_WIH = 76, I_L0_BHH = 77, I_L0_BIH = 78,
  I_L1_WHH = 79, I_L1_WIH = 80, I_L1_BHH = 81, I_L1_BIH = 82,
  I_PN0_W = 83, I_PN0_B = 84, I_PN1_W = 85, I_PN1_B = 86,
  I_PROB_W = 87, I_PROB_B = 88,
  I_ENC = 89, I_TEXTLEN = 90, I_FEAT_T = 91, I_LIP_T = 92,
};
// Leaf offsets inside a conv block dict:
enum { L_B1 = 0, L_B2 = 1, L_BE1 = 2, L_BE2 = 3, L_G1 = 4, L_G2 = 5, L_W1 = 6, L_W2 = 7 };

extern "C" void kernel_launch(void* const* d_in, const int* in_sizes, int n_in,
                              void* d_out, int out_size, void* d_ws, size_t ws_size,
                              hipStream_t stream) {
  (void)in_sizes; (void)n_in; (void)out_size; (void)ws_size;
  auto F = [&](int i) { return (const float*)d_in[i]; };

  char* base = (char*)d_ws;
  size_t off = 0;
  auto alloc = [&](size_t bytes) -> void* {
    void* p = base + off;
    off += (bytes + 255) & ~(size_t)255;
    return p;
  };

  // ---- state block (zeroed every call, contiguous)
  const size_t state_begin = off;
  int*   bar  = (int*)  alloc(2 * sizeof(int));
  bf16*  A1   = (bf16*) alloc((size_t)16 * 1280 * 2);
  bf16*  A2   = (bf16*) alloc((size_t)16 * 2048 * 2);
  float* h0   = (float*)alloc((size_t)8 * 1024 * 4);
  float* c0   = (float*)alloc((size_t)8 * 1024 * 4);
  float* h1   = (float*)alloc((size_t)8 * 1024 * 4);
  float* c1   = (float*)alloc((size_t)8 * 1024 * 4);
  float* cum  = (float*)alloc((size_t)8 * 150 * 4);
  const size_t state_end = off;

  // ---- packed bf16 weights (all NxK; N padded to multiple of 16)
  bf16* Wcomb1  = (bf16*)alloc((size_t)4096 * 1280 * 2);  // [Wih1[:, :256] | Whh1]
  bf16* Wcomb2  = (bf16*)alloc((size_t)4096 * 2048 * 2);  // [Wih2 | Whh2]
  bf16* Wx1     = (bf16*)alloc((size_t)4096 * 256 * 2);   // Wih1[:, 256:512]
  bf16* Wdec    = (bf16*)alloc((size_t)128 * 1024 * 2);   // W_dec^T
  bf16* Wenc    = (bf16*)alloc((size_t)128 * 256 * 2);    // W_enc^T
  bf16* prenW1  = (bf16*)alloc((size_t)256 * 160 * 2);    // prenet0.W^T
  bf16* prenW2  = (bf16*)alloc((size_t)256 * 256 * 2);    // prenet1.W^T
  bf16* catW    = (bf16*)alloc((size_t)256 * 512 * 2);    // concat.W^T
  bf16* featW   = (bf16*)alloc((size_t)160 * 1280 * 2);   // feat_out.W^T
  bf16* probW   = (bf16*)alloc((size_t)16 * 1280 * 2);    // prob_out.W^T (padded)
  bf16* linW    = (bf16*)alloc((size_t)128 * 1280 * 2);   // lip lin.W^T
  float* gbias1 = (float*)alloc(4096 * 4);
  float* gbias2 = (float*)alloc(4096 * 4);

  // ---- activations
  bf16*  encbf   = (bf16*) alloc((size_t)1200 * 256 * 2);
  float* encproj = (float*)alloc((size_t)1200 * 128 * 4);
  bf16*  Apre    = (bf16*) alloc((size_t)400 * 160 * 2);
  bf16*  P1      = (bf16*) alloc((size_t)400 * 256 * 2);
  bf16*  Xcat    = (bf16*) alloc((size_t)400 * 512 * 2);
  bf16*  xbf     = (bf16*) alloc((size_t)400 * 256 * 2);
  float* xpart   = (float*)alloc((size_t)400 * 4096 * 4);
  float* ebuf    = (float*)alloc((size_t)8 * 150 * 4);
  bf16*  hcs     = (bf16*) alloc((size_t)400 * 1280 * 2);
  float* feats   = (float*)alloc((size_t)400 * 160 * 4);
  float* probs   = (float*)alloc((size_t)400 * 2 * 4);
  float* lin     = (float*)alloc((size_t)400 * 128 * 4);
  float* buf0    = (float*)alloc((size_t)400 * 32 * 24 * 24 * 4);
  float* buf1    = (float*)alloc((size_t)400 * 32 * 24 * 24 * 4);

  float* out      = (float*)d_out;
  float* out_feat = out;                     // (8,80,100)
  float* out_lip  = out + 64000;             // (8,1,48,48,50)
  float* out_log  = out + 64000 + 921600;    // (8,100)
  float* out_att  = out + 64000 + 921600 + 800;  // (8,50,150)

  auto g1d = [](size_t n) { return dim3((unsigned)((n + 255) / 256)); };

  // ---- 0) zero state
  k_zero8<<<g1d(state_end - state_begin), 256, 0, stream>>>(
      (uint8_t*)base + state_begin, state_end - state_begin);

  // ---- 1) weight packing: transpose KxN -> bf16 NxK (padded)
  k_transpose_f2b<<<g1d((size_t)128 * 1024), 256, 0, stream>>>(F(I_ATT_WDEC), Wdec, 1024, 128, 128);
  k_transpose_f2b<<<g1d((size_t)128 * 256), 256, 0, stream>>>(F(I_ATT_WENC), Wenc, 256, 128, 128);
  k_transpose_f2b<<<g1d((size_t)256 * 160), 256, 0, stream>>>(F(I_PN0_W), prenW1, 160, 256, 256);
  k_transpose_f2b<<<g1d((size_t)256 * 256), 256, 0, stream>>>(F(I_PN1_W), prenW2, 256, 256, 256);
  k_transpose_f2b<<<g1d((size_t)256 * 512), 256, 0, stream>>>(F(I_CAT_W), catW, 512, 256, 256);
  k_transpose_f2b<<<g1d((size_t)160 * 1280), 256, 0, stream>>>(F(I_FEAT_W), featW, 1280, 160, 160);
  k_transpose_f2b<<<g1d((size_t)16 * 1280), 256, 0, stream>>>(F(I_PROB_W), probW, 1280, 2, 16);
  k_transpose_f2b<<<g1d((size_t)128 * 1280), 256, 0, stream>>>(F(I_LO_LINW), linW, 1280, 128, 128);
  k_f2b<<<g1d(1200 * 256), 256, 0, stream>>>(F(I_ENC), encbf, 1200 * 256);
  // Wcomb1 = [Wih1[:, :256] | Whh1]  (already NxK in torch layout)
  k_copy2d_f2b<<<g1d(4096 * 256), 256, 0, stream>>>(F(I_L0_WIH), 512, 0, Wcomb1, 1280, 0, 4096, 256);
  k_copy2d_f2b<<<g1d((size_t)4096 * 1024), 256, 0, stream>>>(F(I_L0_WHH), 1024, 0, Wcomb1, 1280, 256, 4096, 1024);
  // Wx1 = Wih1[:, 256:512]
  k_copy2d_f2b<<<g1d(4096 * 256), 256, 0, stream>>>(F(I_L0_WIH), 512, 256, Wx1, 256, 0, 4096, 256);
  // Wcomb2 = [Wih2 | Whh2]
  k_copy2d_f2b<<<g1d((size_t)4096 * 1024), 256, 0, stream>>>(F(I_L1_WIH), 1024, 0, Wcomb2, 2048, 0, 4096, 1024);
  k_copy2d_f2b<<<g1d((size_t)4096 * 1024), 256, 0, stream>>>(F(I_L1_WHH), 1024, 0, Wcomb2, 2048, 1024, 4096, 1024);
  k_addv<<<g1d(4096), 256, 0, stream>>>(F(I_L0_BIH), F(I_L0_BHH), gbias1, 4096);
  k_addv<<<g1d(4096), 256, 0, stream>>>(F(I_L1_BIH), F(I_L1_BHH), gbias2, 4096);

  auto gemm = [&](const bf16* A, const bf16* B, float* C, bf16* Cbf,
                  const float* bias, int M, int N, int K, int lda, int ldc,
                  int relu) {
    dim3 g((unsigned)((N + 127) / 128), (unsigned)(M / 16));
    k_gemm<<<g, 256, 0, stream>>>(A, B, C, Cbf, bias, M, N, K, lda, ldc, relu);
  };

  // ---- 2) time-parallel precompute (teacher forcing)
  gemm(encbf, Wenc, encproj, nullptr, F(I_ATT_BENC), 1200, 128, 256, 256, 128, 0);
  k_build_prenet_in<<<g1d(400 * 160), 256, 0, stream>>>(F(I_FEAT_T), Apre);
  gemm(Apre, prenW1, nullptr, P1, F(I_PN0_B), 400, 256, 160, 160, 256, 1);
  gemm(P1, prenW2, nullptr, Xcat, F(I_PN1_B), 400, 256, 256, 256, 512, 1);

  // lip prenet CNN (400 images)
  k_build_lipin<<<g1d((size_t)400 * 2304), 256, 0, stream>>>(F(I_LIP_T), buf0);
  auto LP = [&](int blk, int leaf) { return F(I_LP_BASE + 8 * blk + leaf); };
  k_conv3<<<g1d((size_t)400 * 1 * 48 * 48), 256, 0, stream>>>(buf0, LP(0, L_W1), LP(0, L_B1), LP(0, L_G1), LP(0, L_BE1), buf1, 400, 1, 1, 48, 48, 1);
  k_conv4s2<<<g1d((size_t)400 * 32 * 24 * 24), 256, 0, stream>>>(buf1, LP(0, L_W2), LP(0, L_B2), LP(0, L_G2), LP(0, L_BE2), buf0, 400, 1, 32, 48, 48, 1);
  k_conv3<<<g1d((size_t)400 * 32 * 24 * 24), 256, 0, stream>>>(buf0, LP(1, L_W1), LP(1, L_B1), LP(1, L_G1), LP(1, L_BE1), buf1, 400, 32, 32, 24, 24, 1);
  k_conv4s2<<<g1d((size_t)400 * 64 * 12 * 12), 256, 0, stream>>>(buf1, LP(1, L_W2), LP(1, L_B2), LP(1, L_G2), LP(1, L_BE2), buf0, 400, 32, 64, 24, 24, 1);
  k_conv3<<<g1d((size_t)400 * 64 * 12 * 12), 256, 0, stream>>>(buf0, LP(2, L_W1), LP(2, L_B1), LP(2, L_G1), LP(2, L_BE1), buf1, 400, 64, 64, 12, 12, 1);
  k_conv4s2<<<g1d((size_t)400 * 128 * 6 * 6), 256, 0, stream>>>(buf1, LP(2, L_W2), LP(2, L_B2), LP(2, L_G2), LP(2, L_BE2), buf0, 400, 64, 128, 12, 12, 1);
  k_conv3<<<g1d((size_t)400 * 128 * 6 * 6), 256, 0, stream>>>(buf0, LP(3, L_W1), LP(3, L_B1), LP(3, L_G1), LP(3, L_BE1), buf1, 400, 128, 128, 6, 6, 1);
  k_conv4s2<<<g1d((size_t)400 * 256 * 3 * 3), 256, 0, stream>>>(buf1, LP(3, L_W2), LP(3, L_B2), LP(3, L_G2), LP(3, L_BE2), buf0, 400, 128, 256, 6, 6, 1);
  k_meanpool<<<g1d(400 * 256), 256, 0, stream>>>(buf0, Xcat);

  gemm(Xcat, catW, nullptr, xbf, F(I_CAT_B), 400, 256, 512, 512, 256, 0);
  gemm(xbf, Wx1, xpart, nullptr, gbias1, 400, 4096, 256, 256, 4096, 0);

  // ---- 3) persistent sequential decoder (attention + LSTMs, 50 steps)
  DecArgs da;
  da.enc = F(I_ENC); da.enc_proj = encproj; da.text_len = (const int*)d_in[I_TEXTLEN];
  da.conv_w = F(I_ATT_CONVW); da.W_conv = F(I_ATT_WCONVH); da.vvec = F(I_ATT_V);
  da.Wdec = Wdec; da.Wcomb1 = Wcomb1; da.Wcomb2 = Wcomb2;
  da.xpart = xpart; da.gbias2 = gbias2;
  da.A1 = A1; da.A2 = A2; da.h0 = h0; da.c0 = c0; da.h1 = h1; da.c1 = c1;
  da.cum = cum; da.ebuf = ebuf; da.hcs = hcs; da.att_out = out_att; da.bar = bar;
  k_decoder<<<NWG, 256, 0, stream>>>(da);

  // ---- 4) time-parallel heads
  gemm(hcs, featW, feats, nullptr, nullptr, 400, 160, 1280, 1280, 160, 0);
  gemm(hcs, probW, probs, nullptr, F(I_PROB_B), 400, 2, 1280, 1280, 2, 0);
  gemm(hcs, linW, lin, nullptr, F(I_LO_LINB), 400, 128, 1280, 1280, 128, 0);

  // lip decoder CNN (400 samples)
  k_bcast9<<<g1d((size_t)400 * 128 * 9), 256, 0, stream>>>(lin, buf0);
  auto LO = [&](int blk, int leaf) { return F(I_LO_BASE + 8 * blk + leaf); };
  k_conv3<<<g1d((size_t)400 * 128 * 3 * 3), 256, 0, stream>>>(buf0, LO(0, L_W1), LO(0, L_B1), LO(0, L_G1), LO(0, L_BE1), buf1, 400, 128, 128, 3, 3, 1);
  k_convT4s2<<<g1d((size_t)400 * 64 * 6 * 6), 256, 0, stream>>>(buf1, LO(0, L_W2), LO(0, L_B2), LO(0, L_G2), LO(0, L_BE2), buf0, 400, 128, 64, 3, 3, 1);
  k_conv3<<<g1d((size_t)400 * 64 * 6 * 6), 256, 0, stream>>>(buf0, LO(1, L_W1), LO(1, L_B1), LO(1, L_G1), LO(1, L_BE1), buf1, 400, 64, 64, 6, 6, 1);
  k_convT4s2<<<g1d((size_t)400 * 32 * 12 * 12), 256, 0, stream>>>(buf1, LO(1, L_W2), LO(1, L_B2), LO(1, L_G2), LO(1, L_BE2), buf0, 400, 64, 32, 6, 6, 1);
  k_conv3<<<g1d((size_t)400 * 32 * 12 * 12), 256, 0, stream>>>(buf0, LO(2, L_W1), LO(2, L_B1), LO(2, L_G1), LO(2, L_BE1), buf1, 400, 32, 32, 12, 12, 1);
  k_convT4s2<<<g1d((size_t)400 * 16 * 24 * 24), 256, 0, stream>>>(buf1, LO(2, L_W2), LO(2, L_B2), LO(2, L_G2), LO(2, L_BE2), buf0, 400, 32, 16, 12, 12, 1);
  k_conv3<<<g1d((size_t)400 * 16 * 24 * 24), 256, 0, stream>>>(buf0, LO(3, L_W1), LO(3, L_B1), nullptr, nullptr, buf1, 400, 16, 16, 24, 24, 0);
  k_convT4s2<<<g1d((size_t)400 * 1 * 48 * 48), 256, 0, stream>>>(buf1, LO(3, L_W2), LO(3, L_B2), nullptr, nullptr, buf0, 400, 16, 1, 24, 24, 0);

  // ---- 5) output re-layout
  k_out_feat<<<g1d(64000), 256, 0, stream>>>(feats, out_feat);
  k_out_logit<<<g1d(800), 256, 0, stream>>>(probs, out_log);
  k_out_lip<<<g1d(921600), 256, 0, stream>>>(buf0, out_lip);
}
